// decodeLayer_8985071583711
// MI455X (gfx1250) — matrix operand order repaired and checked by reading, not compiled, not run on hardware
//
#include <hip/hip_runtime.h>

#define BB  64
#define TT  512
#define CC  384
#define HH  6
#define HD  64
#define FFD 1536
#define BT  (BB * TT)

typedef __attribute__((ext_vector_type(16))) __bf16 v16bf;
typedef __attribute__((ext_vector_type(8)))  float  v8f;
typedef int vs4i __attribute__((vector_size(16)));
typedef __attribute__((address_space(1))) vs4i* async_gptr;
typedef __attribute__((address_space(3))) vs4i* async_lptr;

__device__ __forceinline__ __bf16 tobf(float x) { return (__bf16)x; }

#if defined(__gfx1250__) && __has_builtin(__builtin_amdgcn_global_load_async_to_lds_b128)
#define HAVE_ASYNC_LDS 1
#else
#define HAVE_ASYNC_LDS 0
#endif

__device__ __forceinline__ void wait_asynccnt0() {
#if defined(__gfx1250__)
#if __has_builtin(__builtin_amdgcn_s_wait_asynccnt)
  __builtin_amdgcn_s_wait_asynccnt(0);
#else
  asm volatile("s_wait_asynccnt 0" ::: "memory");
#endif
#endif
}

// ---------------------------------------------------------------------------
// Generic GEMM: out[M,N] = act(A[M,K] @ B + bias)
// BMODE 0: B row-major [K,N].  BMODE 1: "headed" layout [N/64][K][64] (H,C,HD).
// OUTBF 1: write bf16 output, else fp32.
// 128 threads = 4 waves; block tile 128x64; per-wave 32x64; K-step 32.
// bf16 WMMA, f32 accumulation. M % 128 == 0, N % 64 == 0, K % 32 == 0.
// ---------------------------------------------------------------------------
template <int RELU, int BMODE, int OUTBF>
__global__ __launch_bounds__(128) void gemm_ws(const float* __restrict__ A,
                                               const float* __restrict__ Bw,
                                               const float* __restrict__ bias,
                                               void* __restrict__ outp,
                                               int M, int N, int K) {
  const int n0   = blockIdx.x * 64;
  const int m0   = blockIdx.y * 128;
  const int tid  = threadIdx.x;
  const int lane = tid & 31;
  const int wave = tid >> 5;
  const int half = lane >> 4;
  const int l16  = lane & 15;

  __shared__ __align__(16) __bf16 Asm[128 * 40];  // 128x32 tile, padded stride
  __shared__ __align__(16) __bf16 Bsm[32 * 72];   // 32x64 tile, padded stride

  v8f acc[2][4] = {};

  const float* bbase = BMODE ? (Bw + (size_t)(n0 >> 6) * (size_t)K * 64) : Bw;

  for (int k0 = 0; k0 < K; k0 += 32) {
    // ---- stage A: one full 32-float row per thread
    {
      const float* ag = A + (size_t)(m0 + tid) * K + k0;
#pragma unroll
      for (int r = 0; r < 32; r++) Asm[tid * 40 + r] = tobf(ag[r]);
      if (k0 + 32 < K) __builtin_prefetch(ag + 32, 0, 0);
    }
    // ---- stage B: 32x64, 16 floats per thread
    {
      int row = tid >> 2;
      int cb  = (tid & 3) * 16;
#pragma unroll
      for (int r = 0; r < 16; r++) {
        float bval = BMODE ? bbase[(size_t)(k0 + row) * 64 + cb + r]
                           : bbase[(size_t)(k0 + row) * N + n0 + cb + r];
        Bsm[row * 72 + cb + r] = tobf(bval);
      }
      if (k0 + 32 < K) {
        const float* bn = BMODE ? &bbase[(size_t)(k0 + 32 + row) * 64 + cb]
                                : &bbase[(size_t)(k0 + 32 + row) * N + n0 + cb];
        __builtin_prefetch(bn, 0, 0);
      }
    }
    __syncthreads();

    // ---- two A fragments (16x32 each) for this wave's 32 rows
    v16bf af[2];
#pragma unroll
    for (int ms = 0; ms < 2; ms++) {
      const int am = wave * 32 + ms * 16 + l16;
#pragma unroll
      for (int e = 0; e < 16; e++) {
        int kk    = (e < 8) ? (half * 8 + e) : (16 + half * 8 + (e - 8));
        af[ms][e] = Asm[am * 40 + kk];
      }
    }
    // ---- 4 B fragments, each feeding 2 WMMAs
#pragma unroll
    for (int nt = 0; nt < 4; nt++) {
      v16bf bfrag;
      int   nn = nt * 16 + l16;
#pragma unroll
      for (int e = 0; e < 16; e++) {
        int kk   = (e < 8) ? (half * 8 + e) : (16 + half * 8 + (e - 8));
        bfrag[e] = Bsm[kk * 72 + nn];
      }
      acc[0][nt] = __builtin_amdgcn_wmma_f32_16x16x32_bf16(
          false, af[0], false, bfrag, (short)0, acc[0][nt], false, false);
      acc[1][nt] = __builtin_amdgcn_wmma_f32_16x16x32_bf16(
          false, af[1], false, bfrag, (short)0, acc[1][nt], false, false);
    }
    __syncthreads();
  }

  // ---- epilogue: bias (+relu); D layout: lane=col l16, VGPR v -> row v+8*half
#pragma unroll
  for (int ms = 0; ms < 2; ms++) {
#pragma unroll
    for (int nt = 0; nt < 4; nt++) {
      int   nn = n0 + nt * 16 + l16;
      float bv = bias[nn];
#pragma unroll
      for (int v = 0; v < 8; v++) {
        int   mm  = m0 + wave * 32 + ms * 16 + v + 8 * half;
        float val = acc[ms][nt][v] + bv;
        if (RELU) val = fmaxf(val, 0.0f);
        if (OUTBF)
          ((__bf16*)outp)[(size_t)mm * N + nn] = tobf(val);
        else
          ((float*)outp)[(size_t)mm * N + nn] = val;
      }
    }
  }
}

// ---------------------------------------------------------------------------
// RoPE (reference style: angle = t for every pair), in-place on bf16 Q and K.
// ---------------------------------------------------------------------------
__global__ void rope_kernel(__bf16* __restrict__ q, __bf16* __restrict__ k) {
  int idx = blockIdx.x * blockDim.x + threadIdx.x;
  if (idx >= BT * (CC / 2)) return;
  int   bt = idx / (CC / 2);
  int   r  = idx % (CC / 2);
  int   h  = r / (HD / 2);
  int   p  = r % (HD / 2);
  int   t  = bt % TT;
  float c  = cosf((float)t), s = sinf((float)t);
  size_t base = (size_t)bt * CC + h * HD + 2 * p;
  float  xr = (float)q[base], xi = (float)q[base + 1];
  q[base]     = tobf(c * xr - s * xi);
  q[base + 1] = tobf(s * xr + c * xi);
  xr = (float)k[base]; xi = (float)k[base + 1];
  k[base]     = tobf(c * xr - s * xi);
  k[base + 1] = tobf(s * xr + c * xi);
}

// ---------------------------------------------------------------------------
// Causal flash attention. grid = (T/64, B*H). 4 waves, 16 query rows each.
// Q/K/V bf16 in [B*T, C] layout at head offset h*HD; O fp32. HD = 64.
// K/V tiles staged to LDS via gfx1250 async global->LDS copies when available.
// ---------------------------------------------------------------------------
__global__ __launch_bounds__(128) void flash_attn(const __bf16* __restrict__ q,
                                                  const __bf16* __restrict__ k,
                                                  const __bf16* __restrict__ v,
                                                  float* __restrict__ o) {
  const int qb   = blockIdx.x * 64;
  const int bh   = blockIdx.y;
  const int b    = bh / HH, h = bh % HH;
  const int hoff = h * HD;
  const int tid  = threadIdx.x, lane = tid & 31, wave = tid >> 5;
  const int half = lane >> 4, l16 = lane & 15;
  const float scale = 0.051031036307982884f;  // 384^-0.5

  __shared__ __align__(16) __bf16 Ksm[32 * 72];
  __shared__ __align__(16) __bf16 Vsm[32 * 72];
  __shared__ __align__(16) __bf16 Psm[4][16 * 40];

  // Q fragments: direct bf16 copy (scale applied to S post-WMMA)
  v16bf aq0, aq1;
  {
    const __bf16* qr = q + (size_t)(b * TT + qb + wave * 16 + l16) * CC + hoff;
#pragma unroll
    for (int e = 0; e < 16; e++) {
      int kk = (e < 8) ? (half * 8 + e) : (16 + half * 8 + (e - 8));
      aq0[e] = qr[kk];
      aq1[e] = qr[32 + kk];
    }
  }

  float rmax[8], rsum[8];
  v8f   oa[4] = {};
#pragma unroll
  for (int vv = 0; vv < 8; vv++) { rmax[vv] = -3.0e38f; rsum[vv] = 0.0f; }

  const int nkb = (qb + 64) / 32;  // causal coverage for this query block
  for (int kb = 0; kb < nkb; kb++) {
    const int kbase = kb * 32;
    // ---- stage K,V block (32x64 bf16): 16B chunks, raw copy (no conversion)
    {
      // chunk i in [0,256): row = i>>3, 16B sub-chunk = i&7 ; thread does i, i+128
#pragma unroll
      for (int rep = 0; rep < 2; rep++) {
        int row = (tid + rep * 128) >> 3;
        int c8  = (tid + rep * 128) & 7;
        const __bf16* kg = k + (size_t)(b * TT + kbase + row) * CC + hoff + c8 * 8;
        const __bf16* vg = v + (size_t)(b * TT + kbase + row) * CC + hoff + c8 * 8;
        __bf16* kl = &Ksm[row * 72 + c8 * 8];
        __bf16* vl = &Vsm[row * 72 + c8 * 8];
#if HAVE_ASYNC_LDS
        __builtin_amdgcn_global_load_async_to_lds_b128((async_gptr)kg,
                                                       (async_lptr)kl, 0, 0);
        __builtin_amdgcn_global_load_async_to_lds_b128((async_gptr)vg,
                                                       (async_lptr)vl, 0, 0);
#else
        *(uint4*)kl = *(const uint4*)kg;
        *(uint4*)vl = *(const uint4*)vg;
#endif
      }
#if HAVE_ASYNC_LDS
      wait_asynccnt0();
#endif
    }
    __syncthreads();

    // ---- S = Q K^T (16 rows x 32 keys): 2 n-tiles x 2 k-chunks
    v8f s0 = {}, s1 = {};
#pragma unroll
    for (int c = 0; c < 2; c++) {
      v16bf bk0, bk1;
#pragma unroll
      for (int e = 0; e < 16; e++) {
        int dd = c * 32 + ((e < 8) ? (half * 8 + e) : (16 + half * 8 + (e - 8)));
        bk0[e] = Ksm[l16 * 72 + dd];
        bk1[e] = Ksm[(16 + l16) * 72 + dd];
      }
      v16bf aqc = c ? aq1 : aq0;
      s0 = __builtin_amdgcn_wmma_f32_16x16x32_bf16(false, aqc, false, bk0,
                                                   (short)0, s0, false, false);
      s1 = __builtin_amdgcn_wmma_f32_16x16x32_bf16(false, aqc, false, bk1,
                                                   (short)0, s1, false, false);
    }

    // ---- causal mask + online softmax (rows live across 16 lanes of a half)
    float p0v[8], p1v[8];
#pragma unroll
    for (int vv = 0; vv < 8; vv++) {
      int   qrow = qb + wave * 16 + vv + 8 * half;
      float a  = s0[vv] * scale;
      float c2 = s1[vv] * scale;
      if (kbase + l16 > qrow) a = -3.0e38f;
      if (kbase + 16 + l16 > qrow) c2 = -3.0e38f;
      float m = fmaxf(a, c2);
#pragma unroll
      for (int off = 1; off < 16; off <<= 1) m = fmaxf(m, __shfl_xor(m, off, 32));
      float mnew = fmaxf(rmax[vv], m);
      float f    = __expf(rmax[vv] - mnew);
      float e0   = __expf(a - mnew);
      float e1   = __expf(c2 - mnew);
      float ps   = e0 + e1;
#pragma unroll
      for (int off = 1; off < 16; off <<= 1) ps += __shfl_xor(ps, off, 32);
      rsum[vv] = rsum[vv] * f + ps;
      rmax[vv] = mnew;
      p0v[vv]  = e0;
      p1v[vv]  = e1;
#pragma unroll
      for (int nt = 0; nt < 4; nt++) oa[nt][vv] *= f;
    }

    // ---- restage P (D layout -> LDS) so it can be re-read in A layout
#pragma unroll
    for (int vv = 0; vv < 8; vv++) {
      int mrow = vv + 8 * half;
      Psm[wave][mrow * 40 + l16]      = tobf(p0v[vv]);
      Psm[wave][mrow * 40 + 16 + l16] = tobf(p1v[vv]);
    }
    __syncthreads();

    v16bf ap;
#pragma unroll
    for (int e = 0; e < 16; e++) {
      int kk = (e < 8) ? (half * 8 + e) : (16 + half * 8 + (e - 8));
      ap[e]  = Psm[wave][l16 * 40 + kk];
    }
    // ---- O += P @ V   (16x32 @ 32x64)
#pragma unroll
    for (int nt = 0; nt < 4; nt++) {
      v16bf bfrag;
#pragma unroll
      for (int e = 0; e < 16; e++) {
        int kk   = (e < 8) ? (half * 8 + e) : (16 + half * 8 + (e - 8));
        bfrag[e] = Vsm[kk * 72 + nt * 16 + l16];
      }
      oa[nt] = __builtin_amdgcn_wmma_f32_16x16x32_bf16(
          false, ap, false, bfrag, (short)0, oa[nt], false, false);
    }
    __syncthreads();
  }

  // ---- normalize + store (fp32)
#pragma unroll
  for (int nt = 0; nt < 4; nt++) {
    int dd = nt * 16 + l16;
#pragma unroll
    for (int vv = 0; vv < 8; vv++) {
      int qrow = qb + wave * 16 + vv + 8 * half;
      o[(size_t)(b * TT + qrow) * CC + hoff + dd] = oa[nt][vv] / rsum[vv];
    }
  }
}

// ---------------------------------------------------------------------------
// out = LN(x + y) * g + b ; one 128-thread block per row of C=384.
// ---------------------------------------------------------------------------
__global__ __launch_bounds__(128) void add_ln(const float* __restrict__ x,
                                              const float* __restrict__ y,
                                              const float* __restrict__ g,
                                              const float* __restrict__ bt,
                                              float* __restrict__ out) {
  const int    row = blockIdx.x;
  const float* xr  = x + (size_t)row * CC;
  const float* yr  = y + (size_t)row * CC;
  __shared__ float red[4];
  float vals[3], s = 0.f;
#pragma unroll
  for (int i = 0; i < 3; i++) {
    int c   = threadIdx.x + i * 128;
    vals[i] = xr[c] + yr[c];
    s += vals[i];
  }
#pragma unroll
  for (int off = 1; off < 32; off <<= 1) s += __shfl_xor(s, off, 32);
  int wave = threadIdx.x >> 5, lane = threadIdx.x & 31;
  if (lane == 0) red[wave] = s;
  __syncthreads();
  float mean = (red[0] + red[1] + red[2] + red[3]) * (1.0f / CC);
  float qv = 0.f;
#pragma unroll
  for (int i = 0; i < 3; i++) { float d = vals[i] - mean; qv += d * d; }
#pragma unroll
  for (int off = 1; off < 32; off <<= 1) qv += __shfl_xor(qv, off, 32);
  __syncthreads();
  if (lane == 0) red[wave] = qv;
  __syncthreads();
  float rstd = rsqrtf((red[0] + red[1] + red[2] + red[3]) * (1.0f / CC) + 1e-5f);
#pragma unroll
  for (int i = 0; i < 3; i++) {
    int c = threadIdx.x + i * 128;
    out[(size_t)row * CC + c] = g[c] * (vals[i] - mean) * rstd + bt[c];
  }
}

// ---------------------------------------------------------------------------
extern "C" void kernel_launch(void* const* d_in, const int* in_sizes, int n_in,
                              void* d_out, int out_size, void* d_ws, size_t ws_size,
                              hipStream_t stream) {
  const float* x    = (const float*)d_in[0];
  const float* q1w  = (const float*)d_in[1];
  const float* q1b  = (const float*)d_in[2];
  const float* k1w  = (const float*)d_in[3];
  const float* k1b  = (const float*)d_in[4];
  const float* v1w  = (const float*)d_in[5];
  const float* v1b  = (const float*)d_in[6];
  const float* p1w  = (const float*)d_in[7];
  const float* p1b  = (const float*)d_in[8];
  const float* ln1g = (const float*)d_in[9];
  const float* ln1b = (const float*)d_in[10];
  const float* q2w  = (const float*)d_in[11];
  const float* q2b  = (const float*)d_in[12];
  const float* k2w  = (const float*)d_in[13];
  const float* k2b  = (const float*)d_in[14];
  const float* v2w  = (const float*)d_in[15];
  const float* v2b  = (const float*)d_in[16];
  const float* p2w  = (const float*)d_in[17];
  const float* p2b  = (const float*)d_in[18];
  const float* ln2g = (const float*)d_in[19];
  const float* ln2b = (const float*)d_in[20];
  const float* f1w  = (const float*)d_in[21];
  const float* f1b  = (const float*)d_in[22];
  const float* f2w  = (const float*)d_in[23];
  const float* f2b  = (const float*)d_in[24];
  const float* ln3g = (const float*)d_in[25];
  const float* ln3b = (const float*)d_in[26];
  float*       out  = (float*)d_out;

  const size_t SL = (size_t)BT * CC;
  float*  Abuf = (float*)d_ws;          // attn output / h2      (SL f32)
  float*  Pbuf = Abuf + SL;             // proj / ffn output     (SL f32)
  float*  Hbuf = Pbuf + SL;             // out1 hidden           (SL f32)
  __bf16* Qbuf = (__bf16*)(Hbuf + SL);  // bf16 QKV (0.5 unit each)
  __bf16* Kbuf = Qbuf + SL;
  __bf16* Vbuf = Kbuf + SL;
  float*  Fbuf = Hbuf;  // BT*FF == 4*SL floats: spans H..V region (dead by FFN)

  const dim3 blk(128);
  const dim3 gC(CC / 64, BT / 128);    // N=384 GEMMs
  const dim3 gF(FFD / 64, BT / 128);   // N=1536 GEMM
  const dim3 gAtt(TT / 64, BB * HH);
  const int  ropeBlocks = (BT * (CC / 2) + 255) / 256;

  // ---- layer 1 attention
  gemm_ws<0, 1, 1><<<gC, blk, 0, stream>>>(x, q1w, q1b, Qbuf, BT, CC, CC);
  gemm_ws<0, 1, 1><<<gC, blk, 0, stream>>>(x, k1w, k1b, Kbuf, BT, CC, CC);
  gemm_ws<0, 1, 1><<<gC, blk, 0, stream>>>(x, v1w, v1b, Vbuf, BT, CC, CC);
  rope_kernel<<<ropeBlocks, 256, 0, stream>>>(Qbuf, Kbuf);
  flash_attn<<<gAtt, blk, 0, stream>>>(Qbuf, Kbuf, Vbuf, Abuf);
  gemm_ws<0, 0, 0><<<gC, blk, 0, stream>>>(Abuf, p1w, p1b, Pbuf, BT, CC, CC);
  add_ln<<<BT, blk, 0, stream>>>(x, Pbuf, ln1g, ln1b, Hbuf);

  // ---- layer 2 attention
  gemm_ws<0, 1, 1><<<gC, blk, 0, stream>>>(Hbuf, q2w, q2b, Qbuf, BT, CC, CC);
  gemm_ws<0, 1, 1><<<gC, blk, 0, stream>>>(Hbuf, k2w, k2b, Kbuf, BT, CC, CC);
  gemm_ws<0, 1, 1><<<gC, blk, 0, stream>>>(Hbuf, v2w, v2b, Vbuf, BT, CC, CC);
  rope_kernel<<<ropeBlocks, 256, 0, stream>>>(Qbuf, Kbuf);
  flash_attn<<<gAtt, blk, 0, stream>>>(Qbuf, Kbuf, Vbuf, Abuf);
  gemm_ws<0, 0, 0><<<gC, blk, 0, stream>>>(Abuf, p2w, p2b, Pbuf, BT, CC, CC);
  add_ln<<<BT, blk, 0, stream>>>(Hbuf, Pbuf, ln2g, ln2b, Abuf);  // Abuf = h2

  // ---- FFN
  gemm_ws<1, 0, 0><<<gF, blk, 0, stream>>>(Abuf, f1w, f1b, Fbuf, BT, FFD, CC);
  gemm_ws<0, 0, 0><<<gC, blk, 0, stream>>>(Fbuf, f2w, f2b, Pbuf, BT, CC, FFD);
  add_ln<<<BT, blk, 0, stream>>>(Abuf, Pbuf, ln3g, ln3b, out);
}
